// ShotNCHead_63591285785127
// MI455X (gfx1250) — matrix operand order repaired and checked by reading, not compiled, not run on hardware
//
#include <hip/hip_runtime.h>
#include <hip/hip_bf16.h>
#include <math.h>

#define N_NODES 50000
#define DD      128
#define CC      64
#define EE      800000
#define INDIM   384
#define EPSF    1e-8f
#define TINYF   1e-12f

typedef __attribute__((ext_vector_type(16))) __bf16          v16bf;
typedef __attribute__((ext_vector_type(8)))  float           v8f;
typedef __attribute__((ext_vector_type(8)))  unsigned short  u16x8;
typedef __attribute__((ext_vector_type(4)))  unsigned short  u16x4;

union BFrag { u16x8 u[2]; v16bf v; };

static __device__ __forceinline__ unsigned short f2bf(float f) {
    unsigned int u = __float_as_uint(f);
    u += 0x7FFFu + ((u >> 16) & 1u);   // round-to-nearest-even
    return (unsigned short)(u >> 16);
}

static __device__ __forceinline__ float waveSum(float v) {
    v += __shfl_xor(v, 16, 32);
    v += __shfl_xor(v,  8, 32);
    v += __shfl_xor(v,  4, 32);
    v += __shfl_xor(v,  2, 32);
    v += __shfl_xor(v,  1, 32);
    return v;
}

// ---------------------------------------------------------------------------
// 1) Per-node log-maps + concat -> bf16 X (N x 384). One wave32 per node.
// ---------------------------------------------------------------------------
__global__ void prep_x(const float* __restrict__ xE, const float* __restrict__ xH,
                       const float* __restrict__ xS, unsigned short* __restrict__ Xb) {
    const int wid  = (blockIdx.x * blockDim.x + threadIdx.x) >> 5;
    const int lane = threadIdx.x & 31;
    if (wid >= N_NODES) return;
    const size_t row = (size_t)wid * DD;
    const int c0 = lane * 4;

    const float4 ev = *(const float4*)(xE + row + c0);
    const float4 hv = *(const float4*)(xH + row + c0);
    const float4 sv = *(const float4*)(xS + row + c0);

    // norms over elements 1..127 (exclude component 0, held by lane 0 slot 0)
    float hp = hv.y*hv.y + hv.z*hv.z + hv.w*hv.w + ((lane == 0) ? 0.0f : hv.x*hv.x);
    float sp = sv.y*sv.y + sv.z*sv.z + sv.w*sv.w + ((lane == 0) ? 0.0f : sv.x*sv.x);
    const float nh = sqrtf(waveSum(hp));
    const float ns = sqrtf(waveSum(sp));
    const float h0 = __shfl(hv.x, 0, 32);
    const float s0 = __shfl(sv.x, 0, 32);
    const float sh = acoshf(fmaxf(h0, 1.0f)) / fmaxf(nh, TINYF);
    const float ss = acosf(fminf(fmaxf(s0, -1.0f), 1.0f)) / fmaxf(ns, TINYF);

    unsigned short* orow = Xb + (size_t)wid * INDIM;
    u16x4 pe = { f2bf(ev.x), f2bf(ev.y), f2bf(ev.z), f2bf(ev.w) };
    *(u16x4*)(orow + c0) = pe;
    const float hx = (lane == 0) ? 0.0f : hv.x * sh;
    u16x4 ph = { f2bf(hx), f2bf(hv.y*sh), f2bf(hv.z*sh), f2bf(hv.w*sh) };
    *(u16x4*)(orow + DD + c0) = ph;
    const float sx = (lane == 0) ? 0.0f : sv.x * ss;
    u16x4 ps = { f2bf(sx), f2bf(sv.y*ss), f2bf(sv.z*ss), f2bf(sv.w*ss) };
    *(u16x4*)(orow + 2*DD + c0) = ps;
}

// ---------------------------------------------------------------------------
// 2) W (384x128 f32, row-major) -> Wt (128x384 bf16, row-major) : B-operand
// ---------------------------------------------------------------------------
__global__ void prep_w(const float* __restrict__ W, unsigned short* __restrict__ Wt) {
    const int t = blockIdx.x * blockDim.x + threadIdx.x;
    if (t >= INDIM * DD) return;
    const int k = t / DD, n = t % DD;
    Wt[(size_t)n * INDIM + k] = f2bf(W[t]);
}

// cls_emb (64x128) -> row-normalized bf16 (64x128). One wave per class.
__global__ void prep_cls(const float* __restrict__ cls, unsigned short* __restrict__ Cn) {
    const int wid  = (blockIdx.x * blockDim.x + threadIdx.x) >> 5;
    const int lane = threadIdx.x & 31;
    if (wid >= CC) return;
    const int c0 = lane * 4;
    const float4 cv = *(const float4*)(cls + (size_t)wid * DD + c0);
    const float nrm = sqrtf(waveSum(cv.x*cv.x + cv.y*cv.y + cv.z*cv.z + cv.w*cv.w));
    const float inv = 1.0f / fmaxf(nrm, EPSF);
    u16x4 p = { f2bf(cv.x*inv), f2bf(cv.y*inv), f2bf(cv.z*inv), f2bf(cv.w*inv) };
    *(u16x4*)(Cn + (size_t)wid * DD + c0) = p;
}

// ---------------------------------------------------------------------------
// 3) workspace init + degrees
// ---------------------------------------------------------------------------
__global__ void init_ws(float* __restrict__ acc, float* __restrict__ deg) {
    const int t = blockIdx.x * blockDim.x + threadIdx.x;
    if (t < N_NODES * DD) acc[t] = 0.0f;
    if (t < N_NODES)      deg[t] = 1.0f;   // self loop
}

__global__ void deg_count(const long long* __restrict__ ei, float* __restrict__ deg) {
    const int e = blockIdx.x * blockDim.x + threadIdx.x;
    if (e < EE) atomicAdd(&deg[(int)ei[EE + e]], 1.0f);
}

__global__ void make_dinv(const float* __restrict__ deg, float* __restrict__ dinv) {
    const int i = blockIdx.x * blockDim.x + threadIdx.x;
    if (i < N_NODES) dinv[i] = rsqrtf(deg[i]);   // deg >= 1 always (self loop)
}

// ---------------------------------------------------------------------------
// 4) H = X(bf16, Nx384) @ Wt(bf16, 128x384 as B) -> f32 Nx128
//    8 waves/block: wave w owns column tile [16w, 16w+16); 12 WMMA over K=384
// ---------------------------------------------------------------------------
__global__ void gemm_xw(const unsigned short* __restrict__ A,
                        const unsigned short* __restrict__ Bt,
                        float* __restrict__ H) {
    const int lane  = threadIdx.x & 31;
    const int wv    = threadIdx.x >> 5;        // 0..7
    const int mbase = blockIdx.x * 16;
    const int nbase = wv * 16;
    const int l15   = lane & 15;
    const int half  = lane >> 4;

    const unsigned short* arow = A  + (size_t)(mbase + l15) * INDIM;
    const unsigned short* brow = Bt + (size_t)(nbase + l15) * INDIM;
    v8f acc = {};
    #pragma unroll
    for (int kb = 0; kb < INDIM; kb += 32) {
        BFrag a, b;
        const unsigned short* ap = arow + kb + 8 * half;   // K: 8h..8h+7 , 16+8h..16+8h+7
        a.u[0] = *(const u16x8*)(ap);
        a.u[1] = *(const u16x8*)(ap + 16);
        const unsigned short* bp = brow + kb + 8 * half;   // K: 8h..8h+7 , 16+8h..16+8h+7
        b.u[0] = *(const u16x8*)(bp);
        b.u[1] = *(const u16x8*)(bp + 16);
        acc = __builtin_amdgcn_wmma_f32_16x16x32_bf16(false, a.v, false, b.v,
                                                      (short)0, acc, false, false);
    }
    float* out = H + (size_t)(mbase + 8 * half) * DD + nbase + l15;
    #pragma unroll
    for (int r = 0; r < 8; ++r) out[(size_t)r * DD] = acc[r];   // m = mbase + r + 8*half
}

// ---------------------------------------------------------------------------
// 5) edge scatter: acc[dst] += h[src] * dinv[src]*dinv[dst]. Wave per edge,
//    2 edges per wave with a prefetch of the second gather row.
// ---------------------------------------------------------------------------
__global__ void scatter_msg(const long long* __restrict__ ei, const float* __restrict__ H,
                            const float* __restrict__ dinv, float* __restrict__ acc) {
    const int wid  = (blockIdx.x * blockDim.x + threadIdx.x) >> 5;
    const int lane = threadIdx.x & 31;
    if (wid >= EE / 2) return;
    const int e0 = wid, e1 = wid + EE / 2;
    const int s0 = (int)ei[e0], d0 = (int)ei[EE + e0];
    const int s1 = (int)ei[e1], d1 = (int)ei[EE + e1];
    __builtin_prefetch(H + (size_t)s1 * DD + lane * 4, 0, 0);  // global_prefetch_b8

    {
        const float nrm = dinv[s0] * dinv[d0];
        const float4 hv = *(const float4*)(H + (size_t)s0 * DD + lane * 4);
        float* ap = acc + (size_t)d0 * DD + lane * 4;
        atomicAdd(ap + 0, hv.x * nrm);
        atomicAdd(ap + 1, hv.y * nrm);
        atomicAdd(ap + 2, hv.z * nrm);
        atomicAdd(ap + 3, hv.w * nrm);
    }
    {
        const float nrm = dinv[s1] * dinv[d1];
        const float4 hv = *(const float4*)(H + (size_t)s1 * DD + lane * 4);
        float* ap = acc + (size_t)d1 * DD + lane * 4;
        atomicAdd(ap + 0, hv.x * nrm);
        atomicAdd(ap + 1, hv.y * nrm);
        atomicAdd(ap + 2, hv.z * nrm);
        atomicAdd(ap + 3, hv.w * nrm);
    }
}

// ---------------------------------------------------------------------------
// 6) finalize: x = acc + h*dinv^2 (self loop) + b ; row-normalize -> bf16 Xn
// ---------------------------------------------------------------------------
__global__ void finalize(const float* __restrict__ H, const float* __restrict__ acc,
                         const float* __restrict__ dinv, const float* __restrict__ bias,
                         unsigned short* __restrict__ Xn) {
    const int wid  = (blockIdx.x * blockDim.x + threadIdx.x) >> 5;
    const int lane = threadIdx.x & 31;
    if (wid >= N_NODES) return;
    const int c0 = lane * 4;
    const float di = dinv[wid];
    const float w  = di * di;
    const float4 av = *(const float4*)(acc  + (size_t)wid * DD + c0);
    const float4 hv = *(const float4*)(H    + (size_t)wid * DD + c0);
    const float4 bv = *(const float4*)(bias + c0);
    const float x0 = av.x + hv.x * w + bv.x;
    const float x1 = av.y + hv.y * w + bv.y;
    const float x2 = av.z + hv.z * w + bv.z;
    const float x3 = av.w + hv.w * w + bv.w;
    const float nrm = sqrtf(waveSum(x0*x0 + x1*x1 + x2*x2 + x3*x3));
    const float inv = 1.0f / fmaxf(nrm, EPSF);
    u16x4 p = { f2bf(x0*inv), f2bf(x1*inv), f2bf(x2*inv), f2bf(x3*inv) };
    *(u16x4*)(Xn + (size_t)wid * DD + c0) = p;
}

// ---------------------------------------------------------------------------
// 7) out = Xn (Nx128 bf16) @ Cn^T (Cn is 64x128 bf16 rows) -> f32 Nx64
//    4 waves/block (4 column tiles of 16 = 64), 4 WMMA over K=128
// ---------------------------------------------------------------------------
__global__ void gemm_out(const unsigned short* __restrict__ A,
                         const unsigned short* __restrict__ Bt,
                         float* __restrict__ O) {
    const int lane  = threadIdx.x & 31;
    const int wv    = threadIdx.x >> 5;        // 0..3
    const int mbase = blockIdx.x * 16;
    const int nbase = wv * 16;
    const int l15   = lane & 15;
    const int half  = lane >> 4;

    const unsigned short* arow = A  + (size_t)(mbase + l15) * DD;
    const unsigned short* brow = Bt + (size_t)(nbase + l15) * DD;
    v8f acc = {};
    #pragma unroll
    for (int kb = 0; kb < DD; kb += 32) {
        BFrag a, b;
        const unsigned short* ap = arow + kb + 8 * half;
        a.u[0] = *(const u16x8*)(ap);
        a.u[1] = *(const u16x8*)(ap + 16);
        const unsigned short* bp = brow + kb + 8 * half;
        b.u[0] = *(const u16x8*)(bp);
        b.u[1] = *(const u16x8*)(bp + 16);
        acc = __builtin_amdgcn_wmma_f32_16x16x32_bf16(false, a.v, false, b.v,
                                                      (short)0, acc, false, false);
    }
    float* out = O + (size_t)(mbase + 8 * half) * CC + nbase + l15;
    #pragma unroll
    for (int r = 0; r < 8; ++r) out[(size_t)r * CC] = acc[r];
}

// ---------------------------------------------------------------------------
extern "C" void kernel_launch(void* const* d_in, const int* in_sizes, int n_in,
                              void* d_out, int out_size, void* d_ws, size_t ws_size,
                              hipStream_t stream) {
    const float*     xE   = (const float*)d_in[0];
    const float*     xH   = (const float*)d_in[1];
    const float*     xS   = (const float*)d_in[2];
    const long long* ei   = (const long long*)d_in[3];   // int64 (2, E)
    const float*     W    = (const float*)d_in[4];
    const float*     bias = (const float*)d_in[5];
    const float*     cls  = (const float*)d_in[6];
    float*           out  = (float*)d_out;

    char* ws = (char*)d_ws;
    size_t off = 0;
    auto carve = [&](size_t bytes) {
        void* p = ws + off;
        off = (off + bytes + 255) & ~(size_t)255;
        return p;
    };
    unsigned short* Xb   = (unsigned short*)carve((size_t)N_NODES * INDIM * 2);
    unsigned short* Wt   = (unsigned short*)carve((size_t)DD * INDIM * 2);
    float*          H    = (float*)carve((size_t)N_NODES * DD * 4);
    float*          ACC  = (float*)carve((size_t)N_NODES * DD * 4);
    float*          DEG  = (float*)carve((size_t)N_NODES * 4);
    float*          DINV = (float*)carve((size_t)N_NODES * 4);
    unsigned short* Xn   = (unsigned short*)carve((size_t)N_NODES * DD * 2);
    unsigned short* Cn   = (unsigned short*)carve((size_t)CC * DD * 2);

    // elementwise prep (independent)
    prep_x  <<<N_NODES / 8, 256, 0, stream>>>(xE, xH, xS, Xb);
    prep_w  <<<(INDIM * DD + 255) / 256, 256, 0, stream>>>(W, Wt);
    prep_cls<<<(CC + 7) / 8, 256, 0, stream>>>(cls, Cn);
    init_ws <<<(N_NODES * DD + 255) / 256, 256, 0, stream>>>(ACC, DEG);
    deg_count<<<(EE + 255) / 256, 256, 0, stream>>>(ei, DEG);
    make_dinv<<<(N_NODES + 255) / 256, 256, 0, stream>>>(DEG, DINV);

    // H = X @ W (WMMA bf16, f32 accum); 8 col tiles x 16 = 128
    gemm_xw<<<N_NODES / 16, 256, 0, stream>>>(Xb, Wt, H);

    // edge scatter (bandwidth-dominant phase)
    scatter_msg<<<(EE / 2) / 8, 256, 0, stream>>>(ei, H, DINV, ACC);

    // self loop + bias + row-normalize -> bf16
    finalize<<<N_NODES / 8, 256, 0, stream>>>(H, ACC, DINV, bias, Xn);

    // out = Xn @ Cn^T (WMMA bf16); 4 col tiles x 16 = 64
    gemm_out<<<N_NODES / 16, 128, 0, stream>>>(Xn, Cn, out);
}
